// RWKV_TimeMix_v7_88605175316639
// MI455X (gfx1250) — matrix operand order repaired and checked by reading, not compiled, not run on hardware
//
#include <hip/hip_runtime.h>
#include <hip/hip_bf16.h>
#include <cstdint>
#include <cstddef>

// ---------------------------------------------------------------------------
// RWKV-v7 TimeMix forward for MI455X (gfx1250, wave32, WMMA 16x16x32 f16)
// B=8 T=2048 C=1024 H=16 N=64 LORA=64
//  - GEMMs: f16 WMMA, f32 accum; weights pre-transposed (N-major, K-contig).
//    B tile staged once per block in LDS (GLOBAL_LOAD_ASYNC_TO_LDS_B128,
//    double buffered, ASYNCcnt-paced); A fragments load as b128 from global.
//  - Scan: 128 blocks (b,h), state column per lane in VGPRs, LDS transpose.
// ---------------------------------------------------------------------------

#define Bdim 8
#define Tdim 2048
#define Cdim 1024
#define Hdim 16
#define Ndim 64
#define Ldim 64
#define BTr  (Bdim * Tdim)                    // 16384 rows
#define BTCn ((size_t)Bdim * Tdim * Cdim)     // 16,777,216 elements
#define BTHn (Bdim * Tdim * Hdim)             // 262,144 heads
#define KSTEP 32

#define DECAY_SCALE 0.6065306597126334f       // exp(-0.5)
#define GN_EPS      (1e-5f * (float)Hdim)     // 1.6e-4
#define NORM_EPS    1e-12f

typedef __attribute__((ext_vector_type(16))) _Float16 v16h;
typedef __attribute__((ext_vector_type(8)))  _Float16 h8;
typedef __attribute__((ext_vector_type(8)))  float    v8f;

// exact parameter type of the async-LDS builtin (from hipcc diagnostic):
// pointer to int __attribute__((vector_size(16))) in AS(1)/AS(3)
typedef int v4i_vs __attribute__((vector_size(4 * sizeof(int))));
typedef __attribute__((address_space(1))) v4i_vs* as1_v4i;
typedef __attribute__((address_space(3))) v4i_vs* as3_v4i;

__device__ __forceinline__ float sigmf(float x) { return 1.0f / (1.0f + __expf(-x)); }

// --- gfx1250 async load-to-LDS path (probe via __has_builtin) --------------
#if __has_builtin(__builtin_amdgcn_global_load_async_to_lds_b128)
#define ASYNC_LDS 1
#endif

#if __has_builtin(__builtin_amdgcn_s_wait_asynccnt)
#define WAIT_ASYNC_1() __builtin_amdgcn_s_wait_asynccnt(1)
#define WAIT_ASYNC_0() __builtin_amdgcn_s_wait_asynccnt(0)
#else
#define WAIT_ASYNC_1() asm volatile("s_wait_asynccnt 0x1" ::: "memory")
#define WAIT_ASYNC_0() asm volatile("s_wait_asynccnt 0x0" ::: "memory")
#endif

// ---------------------------------------------------------------------------
// 1) time-shift + 6 lerps, f32 -> f16 (WMMA A operands)
// ---------------------------------------------------------------------------
__global__ __launch_bounds__(256) void prep_mix_kernel(
    const float* __restrict__ x,
    const float* __restrict__ mr, const float* __restrict__ mk,
    const float* __restrict__ mv, const float* __restrict__ mg,
    const float* __restrict__ ma, const float* __restrict__ md,
    _Float16* __restrict__ xr, _Float16* __restrict__ xk,
    _Float16* __restrict__ xv, _Float16* __restrict__ xg,
    _Float16* __restrict__ xa, _Float16* __restrict__ xd)
{
    size_t i = (size_t)blockIdx.x * blockDim.x + threadIdx.x;
    if (i >= BTCn) return;
    int c = (int)(i & (Cdim - 1));
    int t = (int)((i >> 10) & (Tdim - 1));          // C = 2^10, T = 2^11
    float xc = x[i];
    float xs = (t == 0) ? 0.0f : x[i - Cdim];
    float dx = xs - xc;
    xr[i] = (_Float16)(xc + dx * mr[c]);
    xk[i] = (_Float16)(xc + dx * mk[c]);
    xv[i] = (_Float16)(xc + dx * mv[c]);
    xg[i] = (_Float16)(xc + dx * mg[c]);
    xa[i] = (_Float16)(xc + dx * ma[c]);
    xd[i] = (_Float16)(xc + dx * md[c]);
}

// ---------------------------------------------------------------------------
// 2) f32 (K x N, row-major) -> f16 transposed (N x K, K contiguous)
// ---------------------------------------------------------------------------
__global__ __launch_bounds__(256) void transpose_to_f16_kernel(
    const float* __restrict__ src, _Float16* __restrict__ dst, int K, int N)
{
    int i = blockIdx.x * blockDim.x + threadIdx.x;   // i = n*K + k
    if (i >= K * N) return;
    int k = i % K;
    int n = i / K;
    dst[i] = (_Float16)src[(size_t)k * N + n];
}

// ---------------------------------------------------------------------------
// 3) WMMA GEMM: C[M,Nc] = A[M,Kc] * B[Kc,Nc]; A row-major f16,
//    BT = B transposed (Nc x Kc, K contiguous) f16, acc f32.
//    Block: 256 threads = 8 waves; block tile 256(M) x 64(N);
//    wave tile 32(M) x 64(N): 2 A-frags x 4 B-frags -> 8 WMMA per k-step.
//    B tile (64 cols x 32 K = 4KB) staged in LDS once per block, double
//    buffered via GLOBAL_LOAD_ASYNC_TO_LDS + S_WAIT_ASYNCCNT.
// ---------------------------------------------------------------------------
constexpr int EPI_F32   = 0;   // store f32
constexpr int EPI_F16   = 1;   // store f16 (LoRA intermediate)
constexpr int EPI_SIG   = 2;   // sigmoid -> f32
constexpr int EPI_BSIG  = 3;   // sigmoid(x + bias[n]) -> f32
constexpr int EPI_BTANH = 4;   // tanh(x + bias[n]) -> f32

template <int EPI, int Nc, int Kc>
__global__ __launch_bounds__(256) void gemm_wmma_kernel(
    const _Float16* __restrict__ A, const _Float16* __restrict__ BT,
    void* __restrict__ Cout, const float* __restrict__ bias)
{
    __shared__ _Float16 ldsB[2][64 * KSTEP];   // 2 x 4KB double buffer

    const int tid    = threadIdx.x;
    const int wave   = tid >> 5;
    const int lane   = tid & 31;
    const int half16 = lane >> 4;          // 0: lanes 0-15, 1: lanes 16-31
    const int l16    = lane & 15;
    const int m0     = blockIdx.y * 256 + wave * 32;
    const int n0     = blockIdx.x * 64;

    // cooperative B stage: 256 threads x 16B = 64 rows x 32 halves
    const int srow = tid >> 2;             // 0..63  (column n of B)
    const int sch  = (tid & 3) * 8;        // 0,8,16,24 (K chunk)
    const _Float16* sbase = BT + (size_t)(n0 + srow) * Kc + sch;

    auto stageB = [&](int kb, int buf) {
        const _Float16* src = sbase + kb;
        _Float16* dst = &ldsB[buf][srow * KSTEP + sch];
#if defined(ASYNC_LDS)
        __builtin_amdgcn_global_load_async_to_lds_b128(
            (as1_v4i)(uintptr_t)src,
            (as3_v4i)(uint32_t)(uintptr_t)dst, 0, 0);
#else
        *(h8*)dst = *(const h8*)src;
#endif
    };

    v8f acc[2][4];
    #pragma unroll
    for (int mt = 0; mt < 2; ++mt)
        #pragma unroll
        for (int i = 0; i < 4; ++i)
            #pragma unroll
            for (int e = 0; e < 8; ++e) acc[mt][i][e] = 0.0f;

    const _Float16* arow0 = A + (size_t)(m0 + l16) * Kc;
    const _Float16* arow1 = arow0 + (size_t)16 * Kc;

    stageB(0, 0);

    for (int kb = 0; kb < Kc; kb += KSTEP) {
        const int  buf  = (kb / KSTEP) & 1;
        const bool more = (kb + KSTEP) < Kc;
        if (more) stageB(kb + KSTEP, buf ^ 1);
#if defined(ASYNC_LDS)
        if (more) { WAIT_ASYNC_1(); } else { WAIT_ASYNC_0(); }
#endif
        __syncthreads();                       // B tile visible to all waves

        if (more) {
            __builtin_prefetch(arow0 + kb + 2 * KSTEP, 0, 1);  // global_prefetch_b8
            __builtin_prefetch(arow1 + kb + 2 * KSTEP, 0, 1);
        }

        // ---- A fragments: 16x32 f16: K = kb + half*16 + [0..15] (matches B)
        v16h af0, af1;
        {
            h8 lo0 = *(const h8*)(arow0 + kb + half16 * 16);
            h8 hi0 = *(const h8*)(arow0 + kb + half16 * 16 + 8);
            h8 lo1 = *(const h8*)(arow1 + kb + half16 * 16);
            h8 hi1 = *(const h8*)(arow1 + kb + half16 * 16 + 8);
            #pragma unroll
            for (int e = 0; e < 8; ++e) {
                af0[e] = lo0[e]; af0[8 + e] = hi0[e];
                af1[e] = lo1[e]; af1[8 + e] = hi1[e];
            }
        }
        // ---- B fragments from LDS: lane n = ns*16+l16,
        //      K = kb + half*16 + [0..15] -> 32B contiguous -> ds b128 x2
        #pragma unroll
        for (int ns = 0; ns < 4; ++ns) {
            v16h bf = *(const v16h*)&ldsB[buf][(ns * 16 + l16) * KSTEP + half16 * 16];
            acc[0][ns] = __builtin_amdgcn_wmma_f32_16x16x32_f16(
                false, af0, false, bf, (short)0, acc[0][ns], false, false);
            acc[1][ns] = __builtin_amdgcn_wmma_f32_16x16x32_f16(
                false, af1, false, bf, (short)0, acc[1][ns], false, false);
        }
        __syncthreads();                       // reads done before buf reuse
    }

    // ---- store: C/D 16x16 f32 layout: VGPR v -> M = half*8 + v, N = l16 ----
    #pragma unroll
    for (int mt = 0; mt < 2; ++mt) {
        #pragma unroll
        for (int ns = 0; ns < 4; ++ns) {
            int n = n0 + ns * 16 + l16;
            size_t off0 = (size_t)(m0 + mt * 16 + half16 * 8) * Nc + n;
            float b = 0.0f;
            if constexpr (EPI == EPI_BSIG || EPI == EPI_BTANH) b = bias[n];
            #pragma unroll
            for (int vv = 0; vv < 8; ++vv) {
                float val = acc[mt][ns][vv];
                size_t off = off0 + (size_t)vv * Nc;     // immediate offsets
                if constexpr (EPI == EPI_F32) {
                    ((float*)Cout)[off] = val;
                } else if constexpr (EPI == EPI_F16) {
                    ((_Float16*)Cout)[off] = (_Float16)val;
                } else if constexpr (EPI == EPI_SIG) {
                    ((float*)Cout)[off] = sigmf(val);
                } else if constexpr (EPI == EPI_BSIG) {
                    ((float*)Cout)[off] = sigmf(val + b);
                } else if constexpr (EPI == EPI_BTANH) {
                    ((float*)Cout)[off] = tanhf(val + b);
                }
            }
        }
    }
}

// ---------------------------------------------------------------------------
// 4) per-head fused elementwise: decay = exp(-s*sigmoid(decay_pre)),
//    removal_k = k*rkm, replacement_k = k*lerp(1,iclr,mix),
//    kn = removal_k / max(||removal_k||_2, eps).  One wave per (b,t,h).
// ---------------------------------------------------------------------------
__global__ __launch_bounds__(256) void post_heads_kernel(
    const float* __restrict__ k,    const float* __restrict__ iclr,
    const float* __restrict__ dpre, const float* __restrict__ rkm,
    const float* __restrict__ mix,
    float* __restrict__ decay, float* __restrict__ kr, float* __restrict__ kn)
{
    int wid  = blockIdx.x * 8 + (threadIdx.x >> 5);
    int lane = threadIdx.x & 31;
    int h    = wid & (Hdim - 1);
    int row  = wid >> 4;                       // b*T + t
    size_t base = (size_t)row * Cdim + h * Ndim;
    int    c0 = h * Ndim + lane, c1 = c0 + 32;
    size_t i0 = base + lane,     i1 = i0 + 32;

    float k0 = k[i0], k1 = k[i1];
    float rem0 = k0 * rkm[c0], rem1 = k1 * rkm[c1];
    float ic0 = iclr[i0], ic1 = iclr[i1];
    kr[i0] = k0 * (1.0f + (ic0 - 1.0f) * mix[c0]);
    kr[i1] = k1 * (1.0f + (ic1 - 1.0f) * mix[c1]);
    decay[i0] = __expf(-DECAY_SCALE * sigmf(dpre[i0]));
    decay[i1] = __expf(-DECAY_SCALE * sigmf(dpre[i1]));

    float ss = rem0 * rem0 + rem1 * rem1;
    #pragma unroll
    for (int o = 16; o >= 1; o >>= 1) ss += __shfl_xor(ss, o, 32);
    float dn = fmaxf(sqrtf(ss), NORM_EPS);
    kn[i0] = rem0 / dn;
    kn[i1] = rem1 / dn;
}

// ---------------------------------------------------------------------------
// 5) sequential scan. One block (64 threads = 2 waves) per (b,h).
//    Lane kcol owns state column wkv[:, kcol] in 64 VGPRs.
//      proj[k]   = sum_j (a*kn)[j] * S[j]          (local dot, LDS broadcast)
//      S[j]      = d[j]*S[j] - kn[j]*proj + v[j]*kr[k]
//      out[i]    = sum_k S[i][k]*r[k]              (stride-65 LDS transpose)
// ---------------------------------------------------------------------------
__global__ __launch_bounds__(64) void scan_kernel(
    const float* __restrict__ d_,  const float* __restrict__ a_,
    const float* __restrict__ kn_, const float* __restrict__ kr_,
    const float* __restrict__ v_,  const float* __restrict__ r_,
    float* __restrict__ out)
{
    const int b    = blockIdx.x / Hdim;
    const int h    = blockIdx.x % Hdim;
    const int kcol = threadIdx.x;

    __shared__ float wsh[Ndim], dsh[Ndim], knsh[Ndim], vsh[Ndim];
    __shared__ float red[Ndim][Ndim + 1];

    float S[Ndim];
    #pragma unroll
    for (int j = 0; j < Ndim; ++j) S[j] = 0.0f;

    size_t base = (size_t)(b * Tdim) * Cdim + h * Ndim + kcol;

    for (int t = 0; t < Tdim; ++t) {
        size_t idx = base + (size_t)t * Cdim;
        float dk  = d_[idx],  ak = a_[idx], knk = kn_[idx];
        float krk = kr_[idx], vk = v_[idx], rk  = r_[idx];

        __syncthreads();                      // prev-iter LDS reads done
        wsh[kcol]  = ak * knk;
        dsh[kcol]  = dk;
        knsh[kcol] = knk;
        vsh[kcol]  = vk;
        __syncthreads();

        float proj = 0.0f;
        #pragma unroll
        for (int j = 0; j < Ndim; ++j) proj = fmaf(wsh[j], S[j], proj);
        #pragma unroll
        for (int j = 0; j < Ndim; ++j)
            S[j] = dsh[j] * S[j] - knsh[j] * proj + vsh[j] * krk;

        #pragma unroll
        for (int j = 0; j < Ndim; ++j) red[j][kcol] = S[j] * rk;
        __syncthreads();

        float o = 0.0f;
        #pragma unroll
        for (int j = 0; j < Ndim; ++j) o += red[kcol][j];
        out[idx] = o;
    }
}

// ---------------------------------------------------------------------------
// 6) bonus + GroupNorm + affine + gate -> f16 (A operand of final GEMM).
//    One wave per (b,t,h).
// ---------------------------------------------------------------------------
__global__ __launch_bounds__(256) void gn_gate_kernel(
    const float* __restrict__ sout, const float* __restrict__ r,
    const float* __restrict__ kr,   const float* __restrict__ v,
    const float* __restrict__ gate, const float* __restrict__ bm,
    const float* __restrict__ lnw,  const float* __restrict__ lnb,
    _Float16* __restrict__ yg)
{
    int wid  = blockIdx.x * 8 + (threadIdx.x >> 5);
    int lane = threadIdx.x & 31;
    int h    = wid & (Hdim - 1);
    int row  = wid >> 4;
    size_t base = (size_t)row * Cdim + h * Ndim;
    int    c0 = h * Ndim + lane, c1 = c0 + 32;
    size_t i0 = base + lane,     i1 = i0 + 32;

    // bonus scalar: sum_n r*bm*kr over the head
    float bsum = r[i0] * bm[c0] * kr[i0] + r[i1] * bm[c1] * kr[i1];
    #pragma unroll
    for (int o = 16; o >= 1; o >>= 1) bsum += __shfl_xor(bsum, o, 32);

    float y0 = sout[i0] + bsum * v[i0];
    float y1 = sout[i1] + bsum * v[i1];

    float s = y0 + y1;
    #pragma unroll
    for (int o = 16; o >= 1; o >>= 1) s += __shfl_xor(s, o, 32);
    float mean = s * (1.0f / (float)Ndim);

    float d0 = y0 - mean, d1 = y1 - mean;
    float vs = d0 * d0 + d1 * d1;
    #pragma unroll
    for (int o = 16; o >= 1; o >>= 1) vs += __shfl_xor(vs, o, 32);
    float inv = rsqrtf(vs * (1.0f / (float)Ndim) + GN_EPS);

    yg[i0] = (_Float16)((d0 * inv * lnw[c0] + lnb[c0]) * gate[i0]);
    yg[i1] = (_Float16)((d1 * inv * lnw[c1] + lnb[c1]) * gate[i1]);
}

// ---------------------------------------------------------------------------
// host side
// ---------------------------------------------------------------------------
extern "C" void kernel_launch(void* const* d_in, const int* in_sizes, int n_in,
                              void* d_out, int out_size, void* d_ws, size_t ws_size,
                              hipStream_t stream) {
    (void)in_sizes; (void)n_in; (void)out_size; (void)ws_size;

    const float* x          = (const float*)d_in[0];
    const float* mu_r       = (const float*)d_in[1];
    const float* mu_k       = (const float*)d_in[2];
    const float* mu_v       = (const float*)d_in[3];
    const float* mu_g       = (const float*)d_in[4];
    const float* mu_a       = (const float*)d_in[5];
    const float* mu_d       = (const float*)d_in[6];
    const float* decay_A    = (const float*)d_in[7];
    const float* decay_B    = (const float*)d_in[8];
    const float* decay_bias = (const float*)d_in[9];
    const float* iclr_A     = (const float*)d_in[10];
    const float* iclr_B     = (const float*)d_in[11];
    const float* iclr_bias  = (const float*)d_in[12];
    const float* gate_A     = (const float*)d_in[13];
    const float* gate_B     = (const float*)d_in[14];
    const float* W_r        = (const float*)d_in[15];
    const float* W_k        = (const float*)d_in[16];
    const float* W_v        = (const float*)d_in[17];
    const float* W_o        = (const float*)d_in[18];
    const float* rkm        = (const float*)d_in[19];
    const float* iclr_mix   = (const float*)d_in[20];
    const float* bonus_mult = (const float*)d_in[21];
    const float* ln_w       = (const float*)d_in[22];
    const float* ln_b       = (const float*)d_in[23];

    float* y_out = (float*)d_out;            // output 0: (B,T,C)
    float* v_out = (float*)d_out + BTCn;     // output 1: vprime_0 (B,T,C)

    // bump allocator over workspace
    char*  wsb = (char*)d_ws;
    size_t off = 0;
    auto alloc = [&](size_t bytes) -> void* {
        void* p = wsb + off;
        off += (bytes + 255) & ~(size_t)255;
        return p;
    };

    _Float16* xr16 = (_Float16*)alloc(BTCn * 2);
    _Float16* xk16 = (_Float16*)alloc(BTCn * 2);
    _Float16* xv16 = (_Float16*)alloc(BTCn * 2);
    _Float16* xg16 = (_Float16*)alloc(BTCn * 2);
    _Float16* xa16 = (_Float16*)alloc(BTCn * 2);
    _Float16* xd16 = (_Float16*)alloc(BTCn * 2);

    const size_t CC = (size_t)Cdim * Cdim;
    const size_t CL = (size_t)Cdim * Ldim;
    _Float16* WrT = (_Float16*)alloc(CC * 2);     // all weights: transposed f16
    _Float16* WkT = (_Float16*)alloc(CC * 2);
    _Float16* WvT = (_Float16*)alloc(CC * 2);
    _Float16* WoT = (_Float16*)alloc(CC * 2);
    _Float16* gAT = (_Float16*)alloc(CL * 2);
    _Float16* gBT = (_Float16*)alloc(CL * 2);
    _Float16* iAT = (_Float16*)alloc(CL * 2);
    _Float16* iBT = (_Float16*)alloc(CL * 2);
    _Float16* dAT = (_Float16*)alloc(CL * 2);
    _Float16* dBT = (_Float16*)alloc(CL * 2);
    _Float16* lora16 = (_Float16*)alloc((size_t)BTr * Ldim * 2);

    float* r32    = (float*)alloc(BTCn * 4);
    float* k32    = (float*)alloc(BTCn * 4);
    float* gate32 = (float*)alloc(BTCn * 4);
    float* iclr32 = (float*)alloc(BTCn * 4);
    float* dpre32 = (float*)alloc(BTCn * 4);
    float* decay32= (float*)alloc(BTCn * 4);
    float* kr32   = (float*)alloc(BTCn * 4);
    float* kn32   = (float*)alloc(BTCn * 4);
    float* sout32 = (float*)alloc(BTCn * 4);
    _Float16* yg16 = (_Float16*)alloc(BTCn * 2);

    // --- weight transpose + conversions (one pass, ~20MB; L2-resident) ---
    auto cvtT = [&](const float* s, _Float16* d, int K, int N) {
        int n = K * N;
        transpose_to_f16_kernel<<<(n + 255) / 256, 256, 0, stream>>>(s, d, K, N);
    };
    cvtT(W_r, WrT, Cdim, Cdim);  cvtT(W_k, WkT, Cdim, Cdim);
    cvtT(W_v, WvT, Cdim, Cdim);  cvtT(W_o, WoT, Cdim, Cdim);
    cvtT(gate_A, gAT, Cdim, Ldim); cvtT(gate_B, gBT, Ldim, Cdim);
    cvtT(iclr_A, iAT, Cdim, Ldim); cvtT(iclr_B, iBT, Ldim, Cdim);
    cvtT(decay_A, dAT, Cdim, Ldim); cvtT(decay_B, dBT, Ldim, Cdim);

    // --- time shift + lerps ---
    prep_mix_kernel<<<(unsigned)(BTCn / 256), 256, 0, stream>>>(
        x, mu_r, mu_k, mu_v, mu_g, mu_a, mu_d,
        xr16, xk16, xv16, xg16, xa16, xd16);

    // --- big projections: (16384x1024) @ (1024x1024) ---
    dim3 gBig(Cdim / 64, BTr / 256);     // (16, 64)
    dim3 blk(256);
    gemm_wmma_kernel<EPI_F32, Cdim, Cdim><<<gBig, blk, 0, stream>>>(xr16, WrT, r32,  nullptr);
    gemm_wmma_kernel<EPI_F32, Cdim, Cdim><<<gBig, blk, 0, stream>>>(xk16, WkT, k32,  nullptr);
    gemm_wmma_kernel<EPI_F32, Cdim, Cdim><<<gBig, blk, 0, stream>>>(xv16, WvT, v_out, nullptr);

    // --- LoRA branches: down-proj (N=64) then up-proj (K=64) with fused act ---
    dim3 gLo1(1, BTr / 256);             // (1, 64)
    gemm_wmma_kernel<EPI_F16,  Ldim, Cdim><<<gLo1, blk, 0, stream>>>(xg16, gAT, lora16, nullptr);
    gemm_wmma_kernel<EPI_SIG,  Cdim, Ldim><<<gBig, blk, 0, stream>>>(lora16, gBT, gate32, nullptr);
    gemm_wmma_kernel<EPI_F16,  Ldim, Cdim><<<gLo1, blk, 0, stream>>>(xa16, iAT, lora16, nullptr);
    gemm_wmma_kernel<EPI_BSIG, Cdim, Ldim><<<gBig, blk, 0, stream>>>(lora16, iBT, iclr32, iclr_bias);
    gemm_wmma_kernel<EPI_F16,  Ldim, Cdim><<<gLo1, blk, 0, stream>>>(xd16, dAT, lora16, nullptr);
    gemm_wmma_kernel<EPI_BTANH,Cdim, Ldim><<<gBig, blk, 0, stream>>>(lora16, dBT, dpre32, decay_bias);

    // --- per-head elementwise + L2 normalize ---
    post_heads_kernel<<<BTHn / 8, 256, 0, stream>>>(
        k32, iclr32, dpre32, rkm, iclr_mix, decay32, kr32, kn32);

    // --- sequential scan: 128 blocks (b,h), 64 threads each ---
    scan_kernel<<<Bdim * Hdim, 64, 0, stream>>>(
        decay32, iclr32, kn32, kr32, v_out, r32, sout32);

    // --- bonus + GroupNorm + gate ---
    gn_gate_kernel<<<BTHn / 8, 256, 0, stream>>>(
        sout32, r32, kr32, v_out, gate32, bonus_mult, ln_w, ln_b, yg16);

    // --- output projection ---
    gemm_wmma_kernel<EPI_F32, Cdim, Cdim><<<gBig, blk, 0, stream>>>(yg16, WoT, y_out, nullptr);
}
